// OnlineFlashMTPModel_66331474919982
// MI455X (gfx1250) — matrix operand order repaired and checked by reading, not compiled, not run on hardware
//
#include <hip/hip_runtime.h>
#include <hip/hip_bf16.h>

#define S_ 4096
#define H_ 2048
#define NH_ 16
#define HD_ 128
#define N_ 256
#define BS_ 16
#define W_ 128
#define KV_ 144   // W_ + BS_
#define V_ 32000

typedef __attribute__((ext_vector_type(16))) __bf16 v16bf;
typedef __attribute__((ext_vector_type(8)))  float  v8f;

// round-to-nearest-even fp32 -> bf16 bit pattern
__device__ __forceinline__ unsigned short f2bf(float f) {
    unsigned int u = __builtin_bit_cast(unsigned int, f);
    u += 0x7FFFu + ((u >> 16) & 1u);
    return (unsigned short)(u >> 16);
}

// ---------------------------------------------------------------------------
// fp32 -> bf16 elementwise convert (grid-stride)
// ---------------------------------------------------------------------------
__global__ void convert_f32_bf16_kernel(const float* __restrict__ in,
                                        unsigned short* __restrict__ out,
                                        size_t n) {
    size_t i = (size_t)blockIdx.x * blockDim.x + threadIdx.x;
    size_t stride = (size_t)gridDim.x * blockDim.x;
    for (; i < n; i += stride) out[i] = f2bf(in[i]);
}

// ---------------------------------------------------------------------------
// fp32 (R x C) -> bf16 transposed (C x R), 32x32 LDS tiles, 256 threads
// ---------------------------------------------------------------------------
__global__ __launch_bounds__(256)
void transpose_f32_bf16_kernel(const float* __restrict__ in,
                               unsigned short* __restrict__ out,
                               int R, int Ccols) {
    __shared__ float tile[32][33];
    int bx = blockIdx.x * 32;              // column base (input)
    int by = blockIdx.y * 32;              // row base (input)
    int tx = threadIdx.x & 31, ty = threadIdx.x >> 5;   // ty: 0..7
#pragma unroll
    for (int dy = 0; dy < 32; dy += 8) {
        int r = by + ty + dy, c = bx + tx;
        if (r < R && c < Ccols) tile[ty + dy][tx] = in[(size_t)r * Ccols + c];
    }
    __syncthreads();
#pragma unroll
    for (int dy = 0; dy < 32; dy += 8) {
        int r = bx + ty + dy, c = by + tx; // output is Ccols x R
        if (r < Ccols && c < R) out[(size_t)r * R + c] = f2bf(tile[tx][ty + dy]);
    }
}

// ---------------------------------------------------------------------------
// Embedding gather, writes bf16 rows: noise_bf[(n*BS+j), :] = bf16(E[token])
// ---------------------------------------------------------------------------
__global__ void embed_gather_kernel(const int* __restrict__ input_ids,
                                    const int* __restrict__ anchors,
                                    const unsigned char* __restrict__ keep,
                                    const float* __restrict__ E,
                                    unsigned short* __restrict__ noise_bf) {
    int row = blockIdx.x;                  // 0 .. N_*BS_-1
    int n = row >> 4, j = row & 15;
    int tok = 0;
    if (j == 0 && keep[n]) {
        int a = anchors[n];
        a = a < 0 ? 0 : (a > S_ - 1 ? S_ - 1 : a);
        tok = input_ids[a];
    }
    const float* src = E + (size_t)tok * H_;
    unsigned short* dst = noise_bf + (size_t)row * H_;
    for (int c = threadIdx.x; c < H_; c += blockDim.x) dst[c] = f2bf(src[c]);
}

// ---------------------------------------------------------------------------
// NT GEMM, C(MxN) = A(MxK) * Bt(NxK)^T, bf16 operands, fp32 WMMA accumulate.
// Block tile 128x256, 512 threads = 16 wave32 waves (4M x 4N),
// each wave a 32x64 tile -> acc[2][4] (64 VGPRs, no spills).
// Register-pipelined global->LDS copy, double-buffered LDS, pipeline loop
// manually unrolled x2 so buffer indices are compile-time constants
// (requires nsteps even and >= 2; all launches here have K multiple of 64).
// Outputs: fp32 C (if non-null) and/or bf16 Cb (if non-null).
// ---------------------------------------------------------------------------
#define TM 128
#define TN 256
#define TK 32
#define LDP 40   // padded row stride: 80B rows -> 16B-aligned b128, 0 bank conflicts

__global__ __launch_bounds__(512)
void gemm_nt_bf16_wmma_kernel(const unsigned short* __restrict__ A,
                              const unsigned short* __restrict__ Bt,
                              float* __restrict__ C,
                              unsigned short* __restrict__ Cb,
                              int M, int N, int K) {
    __shared__ __align__(16) unsigned short As[2][TM][LDP];  // [buf][m][k]
    __shared__ __align__(16) unsigned short Bs[2][TN][LDP];  // [buf][n][k]

    const int tid  = threadIdx.x;
    const int lane = tid & 31;
    const int w    = tid >> 5;            // 0..15
    const int wr   = w & 3;               // wave row (M), 4 x 32
    const int wc   = w >> 2;              // wave col (N), 4 x 64
    const int half = lane >> 4;
    const int l15  = lane & 15;
    const int mBase = blockIdx.x * TM;
    const int nBase = blockIdx.y * TN;

    v8f acc[2][4];
    v8f zero = {};
#pragma unroll
    for (int i = 0; i < 2; ++i)
#pragma unroll
        for (int j = 0; j < 4; ++j) acc[i][j] = zero;

    const int nsteps = K >> 5;            // K / TK; even and >= 2 here

    uint4 rA;                             // register staging for next stage
    uint4 rB[2];

    // per-thread copy coordinates
    const int ar  = tid >> 2;             // A: 512 chunks, 1 per thread
    const int ac8 = (tid & 3) << 3;

    // issue global loads of stage s into registers (no wait here)
    auto load_regs = [&](int s) {
        int k0 = s << 5;
        rA = *reinterpret_cast<const uint4*>(
            A + (size_t)(mBase + ar) * K + k0 + ac8);
#pragma unroll
        for (int e = 0; e < 2; ++e) {          // B: 1024 chunks, 2 per thread
            int idx8 = e * 512 + tid;
            int r  = idx8 >> 2;
            int c8 = (idx8 & 3) << 3;
            rB[e] = *reinterpret_cast<const uint4*>(
                Bt + (size_t)(nBase + r) * K + k0 + c8);
        }
    };
    // commit staged registers into LDS buffer (this is where loadcnt waits)
    auto store_regs = [&](int buf) {
        *reinterpret_cast<uint4*>(&As[buf][ar][ac8]) = rA;
#pragma unroll
        for (int e = 0; e < 2; ++e) {
            int idx8 = e * 512 + tid;
            int r  = idx8 >> 2;
            int c8 = (idx8 & 3) << 3;
            *reinterpret_cast<uint4*>(&Bs[buf][r][c8]) = rB[e];
        }
    };
    // B fragment load (ISA 7.12.2: lane n = l15, K = half*8 + {0..7, 16..23})
    auto loadB = [&](int buf, int j) {
        v16bf bf;
        int row = wc * 64 + j * 16 + l15;
        int kb  = half * 8;
#pragma unroll
        for (int t = 0; t < 16; ++t)
            bf[t] = __builtin_bit_cast(__bf16, Bs[buf][row][kb + t + ((t >= 8) ? 8 : 0)]);
        return bf;
    };
    // fragments + 8 WMMAs for one K-step (B frag pipelined one j ahead)
    auto compute = [&](int buf) {
        v16bf afrag[2];
#pragma unroll
        for (int i = 0; i < 2; ++i) {
            int row = wr * 32 + i * 16 + l15;
            int kb  = half * 8;
#pragma unroll
            for (int j = 0; j < 16; ++j) {
                int kk = kb + j + ((j >= 8) ? 8 : 0);
                afrag[i][j] = __builtin_bit_cast(__bf16, As[buf][row][kk]);
            }
        }
        v16bf bf0 = loadB(buf, 0), bf1;
#pragma unroll
        for (int j = 0; j < 4; ++j) {
            if (j < 3) bf1 = loadB(buf, j + 1);  // hide LDS latency behind WMMAs
#pragma unroll
            for (int i = 0; i < 2; ++i)
                acc[i][j] = __builtin_amdgcn_wmma_f32_16x16x32_bf16(
                    false, afrag[i], false, bf0,
                    (short)0, acc[i][j], false, false);
            bf0 = bf1;
        }
    };

    // ---- pipelined main loop, unrolled x2 (constant buffer indices)
    load_regs(0);
    store_regs(0);
    __syncthreads();
    load_regs(1);

    for (int s = 0; s + 2 < nsteps; s += 2) {
        compute(0);
        store_regs(1);            // fixed LDS addresses: direct reg->tile commit
        __syncthreads();
        load_regs(s + 2);
        compute(1);
        store_regs(0);
        __syncthreads();
        load_regs(s + 3);
    }
    compute(0);                   // stage nsteps-2 (nsteps even -> buf 0)
    store_regs(1);
    __syncthreads();
    compute(1);                   // stage nsteps-1

    // ---- epilogue: C/D layout (lane n = l15, row m = r + half*8)
#pragma unroll
    for (int i = 0; i < 2; ++i)
#pragma unroll
        for (int j = 0; j < 4; ++j) {
            int row0 = mBase + wr * 32 + i * 16 + half * 8;
            int col  = nBase + wc * 64 + j * 16 + l15;
            if (C) {
#pragma unroll
                for (int r = 0; r < 8; ++r)
                    C[(size_t)(row0 + r) * N + col] = acc[i][j][r];
            }
            if (Cb) {
#pragma unroll
                for (int r = 0; r < 8; ++r)
                    Cb[(size_t)(row0 + r) * N + col] = f2bf(acc[i][j][r]);
            }
        }
}

// ---------------------------------------------------------------------------
// RoPE, in place on fp32. use_anchor: pos = anchors[row/16]+row%16 else row.
// ---------------------------------------------------------------------------
__global__ void rope_kernel(float* __restrict__ x,
                            const int* __restrict__ anchors,
                            int use_anchor) {
    int row = blockIdx.x;
    float pos = use_anchor ? (float)(anchors[row >> 4] + (row & 15)) : (float)row;
    float* rp = x + (size_t)row * H_;
    const float LN10000 = 9.210340371976184f;
    for (int p = threadIdx.x; p < NH_ * (HD_ / 2); p += blockDim.x) {
        int h = p >> 6, i = p & 63;
        float inv = expf(-((2.0f * (float)i) / (float)HD_) * LN10000);
        float ang = pos * inv;
        float s, c;
        sincosf(ang, &s, &c);
        int col = h * HD_ + 2 * i;
        float x1 = rp[col], x2 = rp[col + 1];
        rp[col]     = x1 * c - x2 * s;
        rp[col + 1] = x1 * s + x2 * c;
    }
}

// ---------------------------------------------------------------------------
// Attention: one 128-thread block per (anchor n, head h). Output bf16.
// ---------------------------------------------------------------------------
__global__ __launch_bounds__(128)
void attn_kernel(const float* __restrict__ q,
                 const float* __restrict__ k_t,
                 const float* __restrict__ v_t,
                 const float* __restrict__ k_n,
                 const float* __restrict__ v_n,
                 const int* __restrict__ anchors,
                 const unsigned char* __restrict__ keep,
                 unsigned short* __restrict__ o_bf) {
    __shared__ float qs[BS_][HD_];
    __shared__ float sc[BS_][KV_];
    int blk = blockIdx.x;
    int n = blk >> 4;
    int h = blk & 15;
    int tid = threadIdx.x;
    int a = anchors[n];
    int kp = keep[n] ? 1 : 0;
    const float scale = 0.08838834764831845f;   // 1/sqrt(128)

    for (int idx = tid; idx < BS_ * HD_; idx += 128) {
        int j = idx >> 7, d = idx & 127;
        qs[j][d] = q[(size_t)(n * BS_ + j) * H_ + h * HD_ + d];
    }
    __syncthreads();

    for (int p = tid; p < BS_ * KV_; p += 128) {
        int qi = p / KV_, kk = p % KV_;
        const float* krow;
        int valid = kp;
        if (kk < W_) {
            int s = a - W_ + kk;
            if (s < 0) { s = 0; valid = 0; }
            krow = k_t + (size_t)s * H_ + h * HD_;
        } else {
            krow = k_n + (size_t)(n * BS_ + (kk - W_)) * H_ + h * HD_;
        }
        float accd = 0.f;
#pragma unroll 8
        for (int d = 0; d < HD_; ++d) accd += qs[qi][d] * krow[d];
        sc[qi][kk] = valid ? accd * scale : -1.0e9f;
    }
    __syncthreads();

    if (tid < BS_) {
        float m = -3.4e38f;
        for (int kk = 0; kk < KV_; ++kk) m = fmaxf(m, sc[tid][kk]);
        float sum = 0.f;
        for (int kk = 0; kk < KV_; ++kk) {
            float e = expf(sc[tid][kk] - m);
            sc[tid][kk] = e;
            sum += e;
        }
        float inv = 1.0f / sum;
        for (int kk = 0; kk < KV_; ++kk) sc[tid][kk] *= inv;
    }
    __syncthreads();

    for (int idx = tid; idx < BS_ * HD_; idx += 128) {
        int qi = idx >> 7, d = idx & 127;
        float accd = 0.f;
        for (int kk = 0; kk < KV_; ++kk) {
            const float* vrow;
            if (kk < W_) {
                int s = a - W_ + kk;
                if (s < 0) s = 0;
                vrow = v_t + (size_t)s * H_ + h * HD_;
            } else {
                vrow = v_n + (size_t)(n * BS_ + (kk - W_)) * H_ + h * HD_;
            }
            accd += sc[qi][kk] * vrow[d];
        }
        o_bf[(size_t)(n * BS_ + qi) * H_ + h * HD_ + d] = f2bf(kp ? accd : 0.f);
    }
}

// ---------------------------------------------------------------------------
extern "C" void kernel_launch(void* const* d_in, const int* in_sizes, int n_in,
                              void* d_out, int out_size, void* d_ws, size_t ws_size,
                              hipStream_t stream) {
    const int*           input_ids = (const int*)d_in[0];
    const int*           anchors   = (const int*)d_in[1];
    const unsigned char* keep      = (const unsigned char*)d_in[2];
    const float*         target_h  = (const float*)d_in[3];
    const float*         E         = (const float*)d_in[4];
    const float*         W_fc      = (const float*)d_in[5];
    const float*         Wq        = (const float*)d_in[6];
    const float*         Wk        = (const float*)d_in[7];
    const float*         Wv        = (const float*)d_in[8];
    const float*         Wo        = (const float*)d_in[9];
    const float*         W_lm      = (const float*)d_in[10];
    float* out = (float*)d_out;

    // ---- workspace carve-up (all sizes 16B multiples)
    char* p = (char*)d_ws;
    auto take = [&](size_t bytes) { char* r = p; p += bytes; return r; };
    const size_t ROWS = 4096;                           // all activations: 4096 rows
    unsigned short* target_bf = (unsigned short*)take((size_t)ROWS * 4096 * 2);
    unsigned short* Wfc_t  = (unsigned short*)take((size_t)H_ * 4096 * 2);   // 2048 x 4096
    unsigned short* Wq_t   = (unsigned short*)take((size_t)H_ * H_ * 2);
    unsigned short* Wk_t   = (unsigned short*)take((size_t)H_ * H_ * 2);
    unsigned short* Wv_t   = (unsigned short*)take((size_t)H_ * H_ * 2);
    unsigned short* Wo_t   = (unsigned short*)take((size_t)H_ * H_ * 2);
    unsigned short* Wlm_t  = (unsigned short*)take((size_t)V_ * H_ * 2);     // 32000 x 2048
    unsigned short* noise_bf = (unsigned short*)take((size_t)ROWS * H_ * 2);
    unsigned short* kv_bf  = (unsigned short*)take((size_t)ROWS * H_ * 2);
    float* qb  = (float*)take((size_t)ROWS * H_ * 4);
    float* k_n = (float*)take((size_t)ROWS * H_ * 4);
    float* v_n = (float*)take((size_t)ROWS * H_ * 4);
    float* k_t = (float*)take((size_t)ROWS * H_ * 4);
    float* v_t = (float*)take((size_t)ROWS * H_ * 4);
    unsigned short* o_bf   = (unsigned short*)take((size_t)ROWS * H_ * 2);
    unsigned short* hid_bf = (unsigned short*)take((size_t)ROWS * H_ * 2);

    dim3 blk(256);
    dim3 gblk(512);

    // ---- one-time prep: bf16 conversions / weight transposes
    convert_f32_bf16_kernel<<<4096, 256, 0, stream>>>(target_h, target_bf,
                                                      (size_t)ROWS * 4096);
    transpose_f32_bf16_kernel<<<dim3(H_ / 32, 4096 / 32), blk, 0, stream>>>(W_fc, Wfc_t, 4096, H_);
    transpose_f32_bf16_kernel<<<dim3(H_ / 32, H_ / 32), blk, 0, stream>>>(Wq, Wq_t, H_, H_);
    transpose_f32_bf16_kernel<<<dim3(H_ / 32, H_ / 32), blk, 0, stream>>>(Wk, Wk_t, H_, H_);
    transpose_f32_bf16_kernel<<<dim3(H_ / 32, H_ / 32), blk, 0, stream>>>(Wv, Wv_t, H_, H_);
    transpose_f32_bf16_kernel<<<dim3(H_ / 32, H_ / 32), blk, 0, stream>>>(Wo, Wo_t, H_, H_);
    transpose_f32_bf16_kernel<<<dim3(V_ / 32, H_ / 32), blk, 0, stream>>>(W_lm, Wlm_t, H_, V_);

    embed_gather_kernel<<<4096, 256, 0, stream>>>(input_ids, anchors, keep, E, noise_bf);

    dim3 gH(4096 / TM, H_ / TN);       // (32, 8)
    dim3 gV(4096 / TM, V_ / TN);       // (32, 125) -- M fastest: W_lm streams once

    // kv_hidden only feeds further GEMMs -> bf16-only output
    gemm_nt_bf16_wmma_kernel<<<gH, gblk, 0, stream>>>(target_bf, Wfc_t, nullptr, kv_bf,
                                                      4096, H_, 2 * H_);
    gemm_nt_bf16_wmma_kernel<<<gH, gblk, 0, stream>>>(noise_bf, Wq_t, qb,  nullptr, 4096, H_, H_);
    gemm_nt_bf16_wmma_kernel<<<gH, gblk, 0, stream>>>(noise_bf, Wk_t, k_n, nullptr, 4096, H_, H_);
    gemm_nt_bf16_wmma_kernel<<<gH, gblk, 0, stream>>>(noise_bf, Wv_t, v_n, nullptr, 4096, H_, H_);
    gemm_nt_bf16_wmma_kernel<<<gH, gblk, 0, stream>>>(kv_bf,    Wk_t, k_t, nullptr, 4096, H_, H_);
    gemm_nt_bf16_wmma_kernel<<<gH, gblk, 0, stream>>>(kv_bf,    Wv_t, v_t, nullptr, 4096, H_, H_);

    rope_kernel<<<4096, 256, 0, stream>>>(qb,  anchors, 1);
    rope_kernel<<<4096, 256, 0, stream>>>(k_n, anchors, 1);
    rope_kernel<<<4096, 256, 0, stream>>>(k_t, anchors, 0);

    attn_kernel<<<N_ * NH_, 128, 0, stream>>>(qb, k_t, v_t, k_n, v_n, anchors, keep, o_bf);

    // hid only feeds the LM head -> bf16-only output
    gemm_nt_bf16_wmma_kernel<<<gH, gblk, 0, stream>>>(o_bf, Wo_t, nullptr, hid_bf, 4096, H_, H_);
    gemm_nt_bf16_wmma_kernel<<<gV, gblk, 0, stream>>>(hid_bf, Wlm_t, out, nullptr, 4096, V_, H_);
}
